// NonLocalAttention_52286931861524
// MI455X (gfx1250) — matrix operand order repaired and checked by reading, not compiled, not run on hardware
//
#include <hip/hip_runtime.h>
#include <hip/hip_bf16.h>
#include <math.h>

typedef __bf16 bf16t;
typedef __attribute__((ext_vector_type(16))) __bf16 v16bf;
typedef __attribute__((ext_vector_type(2)))  __bf16 v2bf;
typedef __attribute__((ext_vector_type(8)))  float  v8f;

#define HW   9216
#define CIN  128
#define CE   64
#define NB   2
#define PT   64     // pixels per conv block
#define QB   64     // query rows per attention block (4 waves x 16)
#define KT   64     // key tile (Bc)
#define OCT  256    // total conv output channels (64 Q + 64 K + 128 V)

union FragB { v16bf v; uint4 q[2]; };
union PackB { v2bf v; unsigned int u; };

// ---------------------------------------------------------------------------
// Stage 0: pack w1|w2|wa into one bf16 weight matrix [256][128] (+bias/slope).
// ---------------------------------------------------------------------------
__global__ __launch_bounds__(256) void pack_weights(
    const float* __restrict__ w1, const float* __restrict__ b1, const float* __restrict__ a1,
    const float* __restrict__ w2, const float* __restrict__ b2, const float* __restrict__ a2,
    const float* __restrict__ wa, const float* __restrict__ ba, const float* __restrict__ aa,
    bf16t* __restrict__ Wb, float* __restrict__ Bs, float* __restrict__ Sl)
{
    const int i = blockIdx.x * 256 + threadIdx.x;      // 256*128 weights
    const int row = i >> 7, col = i & 127;
    float v;
    if (row < 64)       v = w1[row * CIN + col];
    else if (row < 128) v = w2[(row - 64) * CIN + col];
    else                v = wa[(row - 128) * CIN + col];
    Wb[i] = (bf16t)v;
    if (i < OCT) {
        float b, s;
        if (i < 64)       { b = b1[i];       s = a1[0]; }
        else if (i < 128) { b = b2[i - 64];  s = a2[0]; }
        else              { b = ba[i - 128]; s = aa[0]; }
        Bs[i] = b; Sl[i] = s;
    }
}

// ---------------------------------------------------------------------------
// Stage 1: QKV 1x1 conv + bias + PReLU as bf16 WMMA GEMM.
// Block = 4 waves, 64 pixels; each wave: 16 pixels x 256 outs = 64 WMMAs.
//   Q,K -> bf16 [N][HW][64]  (pixel-major), V -> bf16 [N][128][HW] (ch-major)
// ---------------------------------------------------------------------------
__global__ __launch_bounds__(128) void qkv_conv_wmma(
    const float* __restrict__ x, const bf16t* __restrict__ Wb,
    const float* __restrict__ Bs, const float* __restrict__ Sl,
    bf16t* __restrict__ Qb, bf16t* __restrict__ Kb, bf16t* __restrict__ Vb)
{
    __shared__ __align__(16) bf16t xsb[PT][136];       // 64 px x 128 ch, pad 136
    const int nb = blockIdx.x / (HW / PT);
    const int p0 = (blockIdx.x % (HW / PT)) * PT;
    const float* xb = x + (size_t)nb * CIN * HW;

    // stage x tile as bf16 (packed pair stores)
    for (int idx = threadIdx.x; idx < PT * (CIN / 2); idx += 128) {
        const int pp = idx & 63, c2 = (idx >> 6) * 2;
        PackB pk;
        pk.v[0] = (bf16t)xb[(size_t)c2 * HW + p0 + pp];
        pk.v[1] = (bf16t)xb[(size_t)(c2 + 1) * HW + p0 + pp];
        *(unsigned int*)&xsb[pp][c2] = pk.u;
    }
    __syncthreads();

    const int wv = threadIdx.x >> 5, lane = threadIdx.x & 31;
    const int lm = lane & 15, h = lane >> 4;
    const int pw = p0 + wv * 16;                       // wave's first pixel

    // A fragments: 16 pixels x 128 ch (4 K-chunks), loaded once
    FragB af[4];
    #pragma unroll
    for (int kk = 0; kk < 4; ++kk) {
        const bf16t* ap = &xsb[wv * 16 + lm][kk * 32];
        af[kk].q[0] = *(const uint4*)(ap + h * 8);
        af[kk].q[1] = *(const uint4*)(ap + 16 + h * 8);
    }

    #pragma unroll
    for (int ch = 0; ch < 16; ++ch) {
        const int oc = ch * 16 + lm;                   // lane's output channel (col)
        const float bias = Bs[oc], slope = Sl[oc];
        v8f acc = {bias, bias, bias, bias, bias, bias, bias, bias};
        #pragma unroll
        for (int kk = 0; kk < 4; ++kk) {
            FragB wf;                                  // B: col = oc, k in table order
            const bf16t* wp = Wb + (size_t)oc * CIN + kk * 32 + h * 8;
            wf.q[0] = *(const uint4*)(wp);
            wf.q[1] = *(const uint4*)(wp + 16);
            acc = __builtin_amdgcn_wmma_f32_16x16x32_bf16(false, af[kk].v, false, wf.v,
                                                          (short)0, acc, false, false);
        }
        #pragma unroll
        for (int r = 0; r < 8; ++r) acc[r] = (acc[r] >= 0.f) ? acc[r] : slope * acc[r];

        if (ch < 8) {                                  // Q / K, pixel-major
            bf16t* dst = (ch < 4) ? Qb : Kb;
            const int o64 = ((ch & 3) * 16) + lm;
            #pragma unroll
            for (int r = 0; r < 8; ++r)
                dst[((size_t)nb * HW + pw + r + 8 * h) * CE + o64] = (bf16t)acc[r];
        } else {                                       // V, channel-major: 8 px contiguous
            PackB t0, t1, t2, t3;
            t0.v[0] = (bf16t)acc[0]; t0.v[1] = (bf16t)acc[1];
            t1.v[0] = (bf16t)acc[2]; t1.v[1] = (bf16t)acc[3];
            t2.v[0] = (bf16t)acc[4]; t2.v[1] = (bf16t)acc[5];
            t3.v[0] = (bf16t)acc[6]; t3.v[1] = (bf16t)acc[7];
            const uint4 pkt = make_uint4(t0.u, t1.u, t2.u, t3.u);
            *(uint4*)(Vb + ((size_t)nb * CIN + (ch - 8) * 16 + lm) * HW + pw + 8 * h) = pkt;
        }
    }
}

// ---------------------------------------------------------------------------
// Stage 2: flash attention, bf16 WMMA 16x16x32, f32 accum, online softmax.
// 4 waves/block, each wave owns 16 query rows; 144 key tiles of 64.
// Per tile per wave: 24 WMMAs (8 QK^T + 16 PV).
// ---------------------------------------------------------------------------
__global__ __launch_bounds__(128) void flash_attn(
    const bf16t* __restrict__ Qb, const bf16t* __restrict__ Kb,
    const bf16t* __restrict__ Vb, float* __restrict__ out)
{
    __shared__ __align__(16) bf16t sK[KT][72];       // 64 keys x 64 ch (pad 72)
    __shared__ __align__(16) bf16t sV[CIN][72];      // 128 ch x 64 keys (pad 72)
    __shared__ __align__(16) float sS[4][16][76];    // per-wave S stage (pad 76)
    __shared__ __align__(16) bf16t sP[4][16][72];    // per-wave P bf16 (pad 72)
    __shared__ float sRow[4][16];

    const int nb   = blockIdx.x / (HW / QB);
    const int q0   = (blockIdx.x % (HW / QB)) * QB;
    const int wv   = threadIdx.x >> 5;
    const int lane = threadIdx.x & 31;
    const int lm   = lane & 15;
    const int h    = lane >> 4;
    const int qw   = q0 + wv * 16;

    const bf16t* qrow = Qb + ((size_t)nb * HW + qw + lm) * CE;
    FragB qf[2];
    #pragma unroll
    for (int kk = 0; kk < 2; ++kk) {
        qf[kk].q[0] = *(const uint4*)(qrow + kk * 32 + h * 8);
        qf[kk].q[1] = *(const uint4*)(qrow + kk * 32 + 16 + h * 8);
    }

    v8f o[8];
    #pragma unroll
    for (int cc = 0; cc < 8; ++cc) { v8f z = {0,0,0,0,0,0,0,0}; o[cc] = z; }
    float run_m = -INFINITY, run_l = 0.f;

    const bf16t* kbase = Kb + (size_t)nb * HW * CE;
    const bf16t* vbase = Vb + (size_t)nb * CIN * HW;

    for (int kb = 0; kb < HW / KT; ++kb) {
        __syncthreads();                              // prev tile fully consumed
        const bf16t* kg = kbase + (size_t)kb * KT * CE;
        for (int idx = threadIdx.x; idx < KT * 8; idx += 128) {
            const int r = idx >> 3, s = idx & 7;
            *(uint4*)&sK[r][s * 8] = *(const uint4*)(kg + r * CE + s * 8);
        }
        for (int idx = threadIdx.x; idx < CIN * 8; idx += 128) {
            const int c = idx >> 3, s = idx & 7;
            *(uint4*)&sV[c][s * 8] = *(const uint4*)(vbase + (size_t)c * HW + kb * KT + s * 8);
        }
        if (kb + 1 < HW / KT) {                       // global_prefetch_b8 next tiles
            __builtin_prefetch(kg + KT * CE + threadIdx.x * 32, 0, 0);
            __builtin_prefetch(vbase + (size_t)threadIdx.x * HW + (kb + 1) * KT, 0, 0);
        }
        __syncthreads();

        // ---- S = Q x K^T ----
        #pragma unroll
        for (int g = 0; g < 4; ++g) {
            v8f s = {0,0,0,0,0,0,0,0};
            #pragma unroll
            for (int kk = 0; kk < 2; ++kk) {
                FragB kf;
                const bf16t* kp = &sK[g * 16 + lm][kk * 32 + h * 8];
                kf.q[0] = *(const uint4*)(kp);
                kf.q[1] = *(const uint4*)(kp + 16);
                s = __builtin_amdgcn_wmma_f32_16x16x32_bf16(false, qf[kk].v, false, kf.v,
                                                            (short)0, s, false, false);
            }
            #pragma unroll
            for (int r = 0; r < 8; ++r) sS[wv][r + 8 * h][g * 16 + lm] = s[r];
        }

        // ---- online softmax ----
        const float* srow = &sS[wv][lm][h * 32];
        float4 va[8];
        float mloc = -INFINITY;
        #pragma unroll
        for (int k = 0; k < 8; ++k) {
            va[k] = *(const float4*)(srow + 4 * k);
            mloc = fmaxf(mloc, fmaxf(fmaxf(va[k].x, va[k].y), fmaxf(va[k].z, va[k].w)));
        }
        mloc = fmaxf(mloc, __shfl_xor(mloc, 16, 32));
        const float mnew = fmaxf(run_m, mloc);
        const float corr = __expf(run_m - mnew);
        float sloc = 0.f;
        unsigned int* prow = (unsigned int*)&sP[wv][lm][h * 32];
        #pragma unroll
        for (int k = 0; k < 8; ++k) {
            const float e0 = __expf(va[k].x - mnew);
            const float e1 = __expf(va[k].y - mnew);
            const float e2 = __expf(va[k].z - mnew);
            const float e3 = __expf(va[k].w - mnew);
            sloc += (e0 + e1) + (e2 + e3);
            PackB p0; p0.v[0] = (bf16t)e0; p0.v[1] = (bf16t)e1;
            PackB p1; p1.v[0] = (bf16t)e2; p1.v[1] = (bf16t)e3;
            prow[2 * k]     = p0.u;
            prow[2 * k + 1] = p1.u;
        }
        sloc += __shfl_xor(sloc, 16, 32);
        run_l = run_l * corr + sloc;
        run_m = mnew;
        if (h == 0) sRow[wv][lm] = corr;

        // ---- rescale accumulators only when a row max moved ----
        if (__any(corr != 1.0f)) {
            float f[8];
            #pragma unroll
            for (int r = 0; r < 8; ++r) f[r] = sRow[wv][r + 8 * h];
            #pragma unroll
            for (int cc = 0; cc < 8; ++cc)
                #pragma unroll
                for (int r = 0; r < 8; ++r) o[cc][r] *= f[r];
        }

        // ---- O += P x V ----
        FragB pf[2];
        #pragma unroll
        for (int kk = 0; kk < 2; ++kk) {
            pf[kk].q[0] = *(const uint4*)&sP[wv][lm][kk * 32 + h * 8];
            pf[kk].q[1] = *(const uint4*)&sP[wv][lm][kk * 32 + 16 + h * 8];
        }
        #pragma unroll
        for (int cc = 0; cc < 8; ++cc) {
            #pragma unroll
            for (int kk = 0; kk < 2; ++kk) {
                FragB vf;
                const bf16t* vp = &sV[cc * 16 + lm][kk * 32 + h * 8];
                vf.q[0] = *(const uint4*)(vp);
                vf.q[1] = *(const uint4*)(vp + 16);
                o[cc] = __builtin_amdgcn_wmma_f32_16x16x32_bf16(false, pf[kk].v, false, vf.v,
                                                                (short)0, o[cc], false, false);
            }
        }
    }

    // ---- epilogue ----
    if (h == 0) sRow[wv][lm] = 1.f / run_l;
    float fi[8];
    #pragma unroll
    for (int r = 0; r < 8; ++r) fi[r] = sRow[wv][r + 8 * h];
    float* ob = out + (size_t)nb * CIN * HW;
    #pragma unroll
    for (int cc = 0; cc < 8; ++cc) {
        float* op = ob + (size_t)(cc * 16 + lm) * HW + qw;
        #pragma unroll
        for (int r = 0; r < 8; ++r) op[r + 8 * h] = o[cc][r] * fi[r];
    }
}

// ---------------------------------------------------------------------------
extern "C" void kernel_launch(void* const* d_in, const int* in_sizes, int n_in,
                              void* d_out, int out_size, void* d_ws, size_t ws_size,
                              hipStream_t stream)
{
    const float* x  = (const float*)d_in[0];
    const float* w1 = (const float*)d_in[1];
    const float* b1 = (const float*)d_in[2];
    const float* a1 = (const float*)d_in[3];
    const float* w2 = (const float*)d_in[4];
    const float* b2 = (const float*)d_in[5];
    const float* a2 = (const float*)d_in[6];
    const float* wa = (const float*)d_in[7];
    const float* ba = (const float*)d_in[8];
    const float* aa = (const float*)d_in[9];

    bf16t* Qb = (bf16t*)d_ws;                          // [N][HW][64]
    bf16t* Kb = Qb + (size_t)NB * HW * CE;             // [N][HW][64]
    bf16t* Vb = Kb + (size_t)NB * HW * CE;             // [N][128][HW]
    bf16t* Wb = Vb + (size_t)NB * CIN * HW;            // [256][128]
    float* Bs = (float*)(Wb + (size_t)OCT * CIN);      // [256]
    float* Sl = Bs + OCT;                              // [256]

    pack_weights<<<(OCT * CIN) / 256, 256, 0, stream>>>(
        w1, b1, a1, w2, b2, a2, wa, ba, aa, Wb, Bs, Sl);
    qkv_conv_wmma<<<NB * (HW / PT), 128, 0, stream>>>(x, Wb, Bs, Sl, Qb, Kb, Vb);
    flash_attn<<<NB * (HW / QB), 128, 0, stream>>>(Qb, Kb, Vb, (float*)d_out);
}
